// _MHA_V2_40183714021844
// MI455X (gfx1250) — compile-verified, matrix operand order checked by reading, not run on hardware
//
#include <hip/hip_runtime.h>
#include <hip/hip_bf16.h>

// ---------------------------------------------------------------------------
// MHA forward for gfx1250 (MI455X): bf16 WMMA everywhere, fp32 accumulate.
//   H=16 heads, dk=dv=64, D=1024, B=2, L=2048.
// GEMMs register-blocked 64x64 per wave (16 WMMAs / K-step);
// flash attention processes 32 query rows per wave for 2x K/V reuse;
// bool mask pre-packed to bits (1 dword load per row per 32-key block).
// ---------------------------------------------------------------------------

typedef __attribute__((ext_vector_type(16))) __bf16 bf16x16;
typedef __attribute__((ext_vector_type(8)))  __bf16 bf16x8;
typedef __attribute__((ext_vector_type(8)))  float  v8f;

#define HC   16
#define LQ   2048
#define DD   1024
#define DH   64
#define LW   (LQ / 32)   // mask words per row = 64

static __device__ __forceinline__ __bf16 f2bf(float f) {
    unsigned u = __builtin_bit_cast(unsigned, f);
    unsigned r = u + 0x7FFFu + ((u >> 16) & 1u);   // round-to-nearest-even
    unsigned short h = (unsigned short)(r >> 16);
    return __builtin_bit_cast(__bf16, h);
}

static __device__ __forceinline__ v8f wmma_bf16(bf16x16 a, bf16x16 b, v8f c) {
    return __builtin_amdgcn_wmma_f32_16x16x32_bf16(
        /*neg_a=*/false, a, /*neg_b=*/false, b,
        /*c_mod=*/(short)0, c, /*reuse_a=*/false, /*reuse_b=*/false);
}

// A/B fragment loader for v_wmma_f32_16x16x32_bf16.
// ISA 7.12.2 (16-bit A 16x32): lanes 0-15 hold row M=lane with K in
// {0..7, 16..23}; lanes 16-31 hold row M=lane-16 with K in {8..15, 24..31}.
// B (32x16) mirrors with columns per lane. Both are 2x16B contiguous loads
// per lane when the K dimension is contiguous in memory.
static __device__ __forceinline__ bf16x16
load_frag(const __bf16* base, int stride, int row, int kk, int hf) {
    union { bf16x16 v; bf16x8 h[2]; } f;
    const __bf16* p = base + (size_t)row * stride + kk + hf * 8;
    f.h[0] = *(const bf16x8*)(p);
    f.h[1] = *(const bf16x8*)(p + 16);
    return f.v;
}

// ---------------------------------------------------------------------------
__global__ void _mha_cvt_bf16(const float* __restrict__ in,
                              __bf16* __restrict__ out, int n) {
    int i = blockIdx.x * blockDim.x + threadIdx.x;
    if (i < n) out[i] = f2bf(in[i]);
}

// Pack bool mask bytes -> bit words: bit b of word (row, w) = mask[row, w*32+b].
__global__ void _mha_maskpack(const unsigned char* __restrict__ mask,
                              unsigned* __restrict__ bits, int nwords) {
    int i = blockIdx.x * blockDim.x + threadIdx.x;
    if (i < nwords) {
        const unsigned char* p = mask + (size_t)i * 32;
        unsigned w = 0;
#pragma unroll
        for (int b = 0; b < 32; ++b) w |= (p[b] ? 1u : 0u) << b;
        bits[i] = w;
    }
}

// ---------------------------------------------------------------------------
// C[M=4096, N=1024] = A[4096,1024] x W[1024,1024]^T  (Linear: x @ W.T).
// One wave computes a 64x64 block (4x4 WMMA tiles): 8 fragment loads ->
// 16 WMMAs per K-step. MODE is compile-time: single store path per launch.
// MODE 0: store bf16 as [B, H, L, 64]        (q, k)
// MODE 1: store bf16 as [B, H, 64, L]        (v transposed for P*V B-frags)
// MODE 2: store fp32 row-major [4096, 1024]  (final output)
template <int MODE>
__global__ __launch_bounds__(256)
void _mha_gemm(const __bf16* __restrict__ A, const __bf16* __restrict__ W,
               __bf16* __restrict__ outb, float* __restrict__ outf) {
    int wave = (blockIdx.x * blockDim.x + threadIdx.x) >> 5;
    int lane = threadIdx.x & 31;
    int hf = lane >> 4;
    int rc = lane & 15;
    const int N64 = DD / 64;               // 16 column blocks
    int tm = (wave / N64) * 64;            // row block base
    int tn = (wave % N64) * 64;            // col block base

    v8f c[4][4] = {};
    for (int kk = 0; kk < DD; kk += 32) {
        bf16x16 a[4], b[4];
#pragma unroll
        for (int i = 0; i < 4; ++i)
            a[i] = load_frag(A, DD, tm + i * 16 + rc, kk, hf);
#pragma unroll
        for (int j = 0; j < 4; ++j)
            b[j] = load_frag(W, DD, tn + j * 16 + rc, kk, hf);
#pragma unroll
        for (int i = 0; i < 4; ++i)
#pragma unroll
            for (int j = 0; j < 4; ++j)
                c[i][j] = wmma_bf16(a[i], b[j], c[i][j]);
    }

    // C/D layout: lane l, VGPR r -> n = l&15, m = r + 8*(l>>4)
#pragma unroll
    for (int i = 0; i < 4; ++i)
#pragma unroll
        for (int j = 0; j < 4; ++j)
#pragma unroll
            for (int r = 0; r < 8; ++r) {
                int m = tm + i * 16 + r + 8 * hf;   // global row in [0, 4096)
                int n = tn + j * 16 + rc;           // global col in [0, 1024)
                float val = c[i][j][r];
                if constexpr (MODE == 2) {
                    outf[(size_t)m * DD + n] = val;
                } else {
                    int bb = m >> 11;
                    int l  = m & (LQ - 1);
                    int h  = n >> 6;
                    int d  = n & (DH - 1);
                    if constexpr (MODE == 0)
                        outb[((((size_t)bb * HC + h) * LQ) + l) * DH + d] = f2bf(val);
                    else
                        outb[((((size_t)bb * HC + h) * DH) + d) * LQ + l] = f2bf(val);
                }
            }
}

// ---------------------------------------------------------------------------
// Flash attention: one wave (32 threads) per (b, h, 32 query rows).
// Two 16-row query tiles share every K and V fragment (2x reuse).
// qh, kh: [B,H,L,64] bf16 ; vT: [B,H,64,L] bf16 ; mbits: [B,L,L/32] words.
// Writes att: [B, L, H*64] bf16.
__global__ __launch_bounds__(32)
void _mha_attn(const __bf16* __restrict__ qh, const __bf16* __restrict__ kh,
               const __bf16* __restrict__ vT, const unsigned* __restrict__ mbits,
               __bf16* __restrict__ att) {
    __shared__ __attribute__((aligned(16))) __bf16 pls[2][16 * 32];

    int blk = blockIdx.x;
    int qb = blk & (LQ / 32 - 1);          // 64 query blocks of 32 rows
    int h  = (blk >> 6) & (HC - 1);
    int bb = blk >> 10;

    int lane = threadIdx.x;
    int hf = lane >> 4;
    int rc = lane & 15;

    const __bf16* qbase = qh + ((size_t)bb * HC + h) * LQ * DH;
    const __bf16* kbase = kh + ((size_t)bb * HC + h) * LQ * DH;
    const __bf16* vbase = vT + ((size_t)bb * HC + h) * DH * LQ;
    const unsigned* mbase = mbits + (size_t)bb * LQ * LW;

    int q0 = qb * 32;
    bf16x16 aq[2][2];
#pragma unroll
    for (int qi = 0; qi < 2; ++qi) {
        aq[qi][0] = load_frag(qbase, DH, q0 + qi * 16 + rc, 0, hf);
        aq[qi][1] = load_frag(qbase, DH, q0 + qi * 16 + rc, 32, hf);
    }

    v8f o[2][4] = {};
    float mrow[2][8], lrow[2][8];
#pragma unroll
    for (int qi = 0; qi < 2; ++qi)
#pragma unroll
        for (int r = 0; r < 8; ++r) { mrow[qi][r] = -__builtin_inff(); lrow[qi][r] = 0.f; }

    const float SCALE = 0.125f;            // 1/sqrt(64)
    const float NEGINF = -__builtin_inff();

    for (int j0 = 0; j0 < LQ; j0 += 32) {
        // prefetch next key/value tiles (global_prefetch_b8 path)
        if (j0 + 32 < LQ) {
            __builtin_prefetch(kbase + (size_t)(j0 + 32 + lane) * DH, 0, 0);
            __builtin_prefetch(vbase + (size_t)lane * LQ + j0 + 32, 0, 0);
            __builtin_prefetch(vbase + (size_t)(32 + lane) * LQ + j0 + 32, 0, 0);
        }

        // ---- S = Q K^T for 2 q-tiles x 32 keys; K frags shared ----
        v8f s[2][2];
#pragma unroll
        for (int jt = 0; jt < 2; ++jt) {
            int krow = j0 + jt * 16 + rc;
            bf16x16 kb0 = load_frag(kbase, DH, krow, 0, hf);
            bf16x16 kb1 = load_frag(kbase, DH, krow, 32, hf);
#pragma unroll
            for (int qi = 0; qi < 2; ++qi) {
                v8f t = {};
                t = wmma_bf16(aq[qi][0], kb0, t);
                t = wmma_bf16(aq[qi][1], kb1, t);
                s[qi][jt] = t;
            }
        }

        // ---- online softmax per q-tile; stage P to LDS in A-frag order ----
        int jw = j0 >> 5;                  // mask word index for this key block
#pragma unroll
        for (int qi = 0; qi < 2; ++qi) {
            float p0a[8], p1a[8];
#pragma unroll
            for (int r = 0; r < 8; ++r) {
                int gq = q0 + qi * 16 + r + 8 * hf;
                unsigned w = mbase[(size_t)gq * LW + jw];
                float s0 = s[qi][0][r] * SCALE;
                float s1 = s[qi][1][r] * SCALE;
                if ((w >> rc) & 1u)        s0 = NEGINF;
                if ((w >> (16 + rc)) & 1u) s1 = NEGINF;

                float mx = fmaxf(s0, s1);
#pragma unroll
                for (int off = 1; off < 16; off <<= 1)
                    mx = fmaxf(mx, __shfl_xor(mx, off, 16));
                float mnew = fmaxf(mrow[qi][r], mx);

                float alpha = (mrow[qi][r] == NEGINF) ? 0.f : __expf(mrow[qi][r] - mnew);
                float p0 = (s0 == NEGINF) ? 0.f : __expf(s0 - mnew);
                float p1 = (s1 == NEGINF) ? 0.f : __expf(s1 - mnew);

                float ps = p0 + p1;
#pragma unroll
                for (int off = 1; off < 16; off <<= 1)
                    ps += __shfl_xor(ps, off, 16);

                lrow[qi][r] = lrow[qi][r] * alpha + ps;
                mrow[qi][r] = mnew;
#pragma unroll
                for (int nt = 0; nt < 4; ++nt) o[qi][nt][r] *= alpha;
                p0a[r] = p0; p1a[r] = p1;
            }
#pragma unroll
            for (int r = 0; r < 8; ++r) {
                int m = r + 8 * hf;
                pls[qi][m * 32 + rc]      = f2bf(p0a[r]);
                pls[qi][m * 32 + 16 + rc] = f2bf(p1a[r]);
            }
        }
        __syncthreads();

        union { bf16x16 v; bf16x8 h[2]; } ap0, ap1;
        ap0.h[0] = *(const bf16x8*)&pls[0][rc * 32 + hf * 8];
        ap0.h[1] = *(const bf16x8*)&pls[0][rc * 32 + 16 + hf * 8];
        ap1.h[0] = *(const bf16x8*)&pls[1][rc * 32 + hf * 8];
        ap1.h[1] = *(const bf16x8*)&pls[1][rc * 32 + 16 + hf * 8];
        __syncthreads();

        // ---- O += P[16x32] * V[32x64]; V frags shared by both q-tiles ----
#pragma unroll
        for (int nt = 0; nt < 4; ++nt) {
            bf16x16 vf = load_frag(vbase, LQ, nt * 16 + rc, j0, hf);
            o[0][nt] = wmma_bf16(ap0.v, vf, o[0][nt]);
            o[1][nt] = wmma_bf16(ap1.v, vf, o[1][nt]);
        }
    }

    // ---- normalize and store att[b, l, h*64 + d] ----
#pragma unroll
    for (int qi = 0; qi < 2; ++qi)
#pragma unroll
        for (int r = 0; r < 8; ++r) {
            int gq = q0 + qi * 16 + r + 8 * hf;
            float inv = (lrow[qi][r] > 0.f) ? (1.f / lrow[qi][r]) : 0.f;
            size_t base = ((size_t)bb * LQ + gq) * DD + (size_t)h * DH + rc;
            att[base + 0]  = f2bf(o[qi][0][r] * inv);
            att[base + 16] = f2bf(o[qi][1][r] * inv);
            att[base + 32] = f2bf(o[qi][2][r] * inv);
            att[base + 48] = f2bf(o[qi][3][r] * inv);
        }
}

// ---------------------------------------------------------------------------
extern "C" void kernel_launch(void* const* d_in, const int* in_sizes, int n_in,
                              void* d_out, int out_size, void* d_ws, size_t ws_size,
                              hipStream_t stream) {
    const float*         queries = (const float*)d_in[0];
    const unsigned char* mask    = (const unsigned char*)d_in[1]; // bool tensor
    const float*         Wq      = (const float*)d_in[2];
    const float*         Wk      = (const float*)d_in[3];
    const float*         Wv      = (const float*)d_in[4];
    const float*         Wo      = (const float*)d_in[5];
    float*               out     = (float*)d_out;

    const int M  = 2 * LQ;                 // 4096 rows
    const int NX = M * DD;                 // 4,194,304 activation elems
    const int NW = DD * DD;                // 1,048,576 weight elems
    const int NH = 2 * HC * LQ * DH;       // 4,194,304 per-head elems
    const int NMW = 2 * LQ * LW;           // 262,144 packed mask words

    __bf16* Xbf  = (__bf16*)d_ws;          // ws layout (bf16 elems), ~49 MB total
    __bf16* Wqb  = Xbf + NX;
    __bf16* Wkb  = Wqb + NW;
    __bf16* Wvb  = Wkb + NW;
    __bf16* Wob  = Wvb + NW;
    __bf16* qhb  = Wob + NW;               // [B,H,L,64]
    __bf16* khb  = qhb + NH;               // [B,H,L,64]
    __bf16* vTb  = khb + NH;               // [B,H,64,L]
    __bf16* attb = vTb + NH;               // [B,L,H*64]
    unsigned* mbits = (unsigned*)(attb + NX);

    // 1) fp32 -> bf16 conversions + mask bit-pack
    _mha_cvt_bf16<<<NX / 256, 256, 0, stream>>>(queries, Xbf, NX);
    _mha_cvt_bf16<<<NW / 256, 256, 0, stream>>>(Wq, Wqb, NW);
    _mha_cvt_bf16<<<NW / 256, 256, 0, stream>>>(Wk, Wkb, NW);
    _mha_cvt_bf16<<<NW / 256, 256, 0, stream>>>(Wv, Wvb, NW);
    _mha_cvt_bf16<<<NW / 256, 256, 0, stream>>>(Wo, Wob, NW);
    _mha_maskpack<<<NMW / 256, 256, 0, stream>>>(mask, mbits, NMW);

    // 2) projections: (M/64)*(D/64) = 1024 waves, 8 waves/block -> 128 blocks
    const int gemm_blocks = (M / 64) * (DD / 64) / 8;
    _mha_gemm<0><<<gemm_blocks, 256, 0, stream>>>(Xbf, Wqb, qhb, nullptr);
    _mha_gemm<0><<<gemm_blocks, 256, 0, stream>>>(Xbf, Wkb, khb, nullptr);
    _mha_gemm<1><<<gemm_blocks, 256, 0, stream>>>(Xbf, Wvb, vTb, nullptr);

    // 3) flash attention: B*H*(L/32) = 2048 waves
    _mha_attn<<<2 * HC * (LQ / 32), 32, 0, stream>>>(qhb, khb, vTb, mbits, attb);

    // 4) output projection -> fp32 d_out
    _mha_gemm<2><<<gemm_blocks, 256, 0, stream>>>(attb, Wob, nullptr, out);
}
